// GATModel_9844065042756
// MI455X (gfx1250) — matrix operand order repaired and checked by reading, not compiled, not run on hardware
//
#include <hip/hip_runtime.h>
#include <hip/hip_bf16.h>
#include <math.h>

typedef __attribute__((ext_vector_type(16))) _Float16 v16h;
typedef __attribute__((ext_vector_type(8)))  _Float16 v8h;
typedef __attribute__((ext_vector_type(8)))  float    v8f;

#define GRID_STRIDE(i, n) \
  for (size_t i = (size_t)blockIdx.x * blockDim.x + threadIdx.x; i < (n); \
       i += (size_t)gridDim.x * blockDim.x)

static inline unsigned blocks_for(size_t n, int bs) {
  size_t b = (n + (size_t)bs - 1) / (size_t)bs;
  if (b > (size_t)1048576) b = 1048576;
  if (b == 0) b = 1;
  return (unsigned)b;
}

// ---------------------------------------------------------------------------
// Generic fill
// ---------------------------------------------------------------------------
__global__ void fill_kernel(float* __restrict__ p, float v, size_t n) {
  GRID_STRIDE(i, n) p[i] = v;
}

// ---------------------------------------------------------------------------
// WMMA GEMM: C[M,N] = A[M,K] @ B[K,N], fp32 in/out, f16 WMMA, f32 accumulate.
// Requires M%16==0, N%64==0, K%32==0.
//
//  * B is staged once per block into LDS as f16, column-major with a padded
//    stride KS=K+8 halves: fragment reads become two ds_load_b128 per subtile
//    with lanes 4 banks apart (conflict-free on the 64-bank LDS).
//  * One wave computes a 16x64 C tile (4 accumulators): the A fragment is
//    loaded once per k-step (4x global_load_b128) and reused by 4 WMMAs.
//  * All four B fragments are loaded into distinct registers BEFORE the WMMAs
//    so the ds_loads clause together behind a single s_wait_dscnt and the
//    four v_wmma issue back-to-back.
//
// Fragment layouts per CDNA5 ISA 7.12.2 (wave32):
//   A (16x32 f16): lane l holds row m=l&15; half j holds K = 16*(j>>3)+8*(l>>4)+(j&7)
//   B (32x16 f16): lane l holds col n=l&15; half j holds K = 16*(j>>3)+8*(l>>4)+(j&7)
//   C (16x16 f32): VGPR r, lane l -> C[r + 8*(l>>4)][l&15]
// ---------------------------------------------------------------------------
__global__ void gemm_wmma_f16(const float* __restrict__ A,
                              const float* __restrict__ B,
                              float* __restrict__ C,
                              int M, int N, int K) {
  extern __shared__ _Float16 Bs[];          // [N][K+8] column-major, padded
  const int KS = K + 8;

  // Cooperative stage: coalesced global reads, scattered LDS writes.
  for (int i = threadIdx.x; i < K * N; i += blockDim.x) {
    const int k = i / N, n = i % N;
    Bs[n * KS + k] = (_Float16)B[i];
  }
  __syncthreads();

  const int lane    = threadIdx.x & 31;
  const int nColGrp = N >> 6;               // 64-wide column groups
  const int wave    = blockIdx.x * (blockDim.x >> 5) + ((int)threadIdx.x >> 5);
  const int rowTile = wave / nColGrp;
  const int colGrp  = wave % nColGrp;
  if (rowTile >= (M >> 4)) return;          // wave-uniform exit (after barrier)

  const int l15 = lane & 15;
  const int hi  = lane >> 4;                // 0 or 1
  const size_t rowA   = (size_t)rowTile * 16 + l15;
  const int   colBase = colGrp * 64;

  v8f acc0 = {}, acc1 = {}, acc2 = {}, acc3 = {};

#pragma unroll 2
  for (int k0 = 0; k0 < K; k0 += 32) {
    // ---- A fragment: two contiguous 8-float runs -> 4x global_load_b128 ----
    float af[16];
    const float* Ap = A + rowA * (size_t)K + k0;
    *(float4*)(af + 0)  = *(const float4*)(Ap + 8 * hi);
    *(float4*)(af + 4)  = *(const float4*)(Ap + 8 * hi + 4);
    *(float4*)(af + 8)  = *(const float4*)(Ap + 16 + 8 * hi);
    *(float4*)(af + 12) = *(const float4*)(Ap + 16 + 8 * hi + 4);
    v16h a;
#pragma unroll
    for (int j = 0; j < 16; ++j) a[j] = (_Float16)af[j];

    // ---- All four B fragments first (8x ds_load_b128, one wait) ----
    const int kb = k0 + 8 * hi;
    v16h bf[4];
#pragma unroll
    for (int s = 0; s < 4; ++s) {
      const _Float16* bp = &Bs[(colBase + s * 16 + l15) * KS + kb];
      const v8h blo = *(const v8h*)(bp);
      const v8h bhi = *(const v8h*)(bp + 16);
#pragma unroll
      for (int j = 0; j < 8; ++j) { bf[s][j] = blo[j]; bf[s][8 + j] = bhi[j]; }
    }

    // ---- Four back-to-back WMMAs on independent accumulators ----
    acc0 = __builtin_amdgcn_wmma_f32_16x16x32_f16(false, a, false, bf[0], (short)0, acc0, false, false);
    acc1 = __builtin_amdgcn_wmma_f32_16x16x32_f16(false, a, false, bf[1], (short)0, acc1, false, false);
    acc2 = __builtin_amdgcn_wmma_f32_16x16x32_f16(false, a, false, bf[2], (short)0, acc2, false, false);
    acc3 = __builtin_amdgcn_wmma_f32_16x16x32_f16(false, a, false, bf[3], (short)0, acc3, false, false);
  }

#pragma unroll
  for (int r = 0; r < 8; ++r) {
    const size_t row = (size_t)(rowTile * 16 + r + 8 * hi);
    float* cp = C + row * (size_t)N + colBase + l15;
    cp[0]  = acc0[r];
    cp[16] = acc1[r];
    cp[32] = acc2[r];
    cp[48] = acc3[r];
  }
}

// ---------------------------------------------------------------------------
// Attention coefficients: as[n,h] = sum_c h[n,h,c]*att_s[h,c]; same for ad.
// ---------------------------------------------------------------------------
__global__ void attn_coef_kernel(const float* __restrict__ h,
                                 const float* __restrict__ att_s,
                                 const float* __restrict__ att_d,
                                 float* __restrict__ as_out,
                                 float* __restrict__ ad_out,
                                 int Nn, int H, int C) {
  GRID_STRIDE(i, (size_t)Nn * H) {
    const int hd = (int)(i % H);
    const size_t n = i / H;
    const float* hp = h + (n * (size_t)H + hd) * (size_t)C;
    const float* sp = att_s + (size_t)hd * C;
    const float* dp = att_d + (size_t)hd * C;
    float s = 0.f, d = 0.f;
    for (int c = 0; c < C; ++c) { s += hp[c] * sp[c]; d += hp[c] * dp[c]; }
    as_out[i] = s;
    ad_out[i] = d;
  }
}

// ---------------------------------------------------------------------------
// Float atomic max via int/uint ordering trick (valid for all signs, init -inf)
// ---------------------------------------------------------------------------
__device__ __forceinline__ void atomicMaxFloat(float* addr, float val) {
  if (val >= 0.f) atomicMax((int*)addr, __float_as_int(val));
  else            atomicMin((unsigned int*)addr, __float_as_uint(val));
}

__device__ __forceinline__ void edge_src_dst(const int* __restrict__ ei,
                                             int E, size_t e, int& src, int& dst) {
  if (e < (size_t)E) { src = ei[e]; dst = ei[(size_t)E + e]; }
  else               { src = dst = (int)(e - (size_t)E); }   // appended self loops
}

// Pass 1: logits e = leaky_relu(as[src]+ad[dst]); store; segment-max into m[dst]
__global__ void edge_logits_max_kernel(const int* __restrict__ ei, int E, int Etot,
                                       const float* __restrict__ as_in,
                                       const float* __restrict__ ad_in,
                                       float* __restrict__ ew,
                                       float* __restrict__ m, int H) {
  GRID_STRIDE(i, (size_t)Etot * H) {
    const int hd = (int)(i % H);
    const size_t e = i / H;
    int src, dst;
    edge_src_dst(ei, E, e, src, dst);
    float v = as_in[(size_t)src * H + hd] + ad_in[(size_t)dst * H + hd];
    v = (v > 0.f) ? v : 0.2f * v;            // negative_slope = 0.2
    ew[i] = v;
    atomicMaxFloat(&m[(size_t)dst * H + hd], v);
  }
}

// Pass 2: p = exp(e - m[dst]); store back; segment-sum into den[dst]
__global__ void edge_exp_sum_kernel(const int* __restrict__ ei, int E, int Etot,
                                    const float* __restrict__ m,
                                    float* __restrict__ ew,
                                    float* __restrict__ den, int H) {
  GRID_STRIDE(i, (size_t)Etot * H) {
    const int hd = (int)(i % H);
    const size_t e = i / H;
    int src, dst;
    edge_src_dst(ei, E, e, src, dst);
    const float p = __expf(ew[i] - m[(size_t)dst * H + hd]);
    ew[i] = p;
    atomicAdd(&den[(size_t)dst * H + hd], p);
  }
}

// Pass 3: out[dst,h,c] += h_in[src,h,c] * (p / den[dst,h]).
// One thread per (edge, head, 4-channel group): float4 gather + 4 atomics,
// channel-fastest indexing keeps gathers and atomics coalesced (L2-resident).
__global__ void edge_agg_kernel(const int* __restrict__ ei, int E, int Etot,
                                const float* __restrict__ ew,
                                const float* __restrict__ den,
                                const float* __restrict__ h_in,
                                float* __restrict__ out, int H, int C) {
  const int Cq = C >> 2;
  const size_t HC = (size_t)H * C;
  GRID_STRIDE(i, (size_t)Etot * H * Cq) {
    const int c  = ((int)(i % Cq)) << 2;
    const int hd = (int)((i / Cq) % H);
    const size_t e = i / ((size_t)Cq * H);
    int src, dst;
    edge_src_dst(ei, E, e, src, dst);
    const float alpha = ew[e * H + hd] / den[(size_t)dst * H + hd];
    const float4 v = *(const float4*)&h_in[(size_t)src * HC + (size_t)hd * C + c];
    float* base = &out[(size_t)dst * HC + (size_t)hd * C + c];
    atomicAdd(base + 0, v.x * alpha);
    atomicAdd(base + 1, v.y * alpha);
    atomicAdd(base + 2, v.z * alpha);
    atomicAdd(base + 3, v.w * alpha);
  }
}

// Bias (+ optional ReLU), in place
__global__ void bias_act_kernel(float* __restrict__ io, const float* __restrict__ bias,
                                int HC, size_t n, int do_relu) {
  GRID_STRIDE(i, n) {
    float v = io[i] + bias[i % (size_t)HC];
    if (do_relu) v = v > 0.f ? v : 0.f;
    io[i] = v;
  }
}

// ---------------------------------------------------------------------------
// Host orchestration
// ---------------------------------------------------------------------------
extern "C" void kernel_launch(void* const* d_in, const int* in_sizes, int n_in,
                              void* d_out, int out_size, void* d_ws, size_t ws_size,
                              hipStream_t stream) {
  (void)n_in; (void)out_size; (void)ws_size;
  const float* x        = (const float*)d_in[0];
  const int*   ei       = (const int*)  d_in[1];
  const float* W1       = (const float*)d_in[2];
  const float* att_src1 = (const float*)d_in[3];
  const float* att_dst1 = (const float*)d_in[4];
  const float* b1       = (const float*)d_in[5];
  const float* W2       = (const float*)d_in[6];
  const float* att_src2 = (const float*)d_in[7];
  const float* att_dst2 = (const float*)d_in[8];
  const float* b2       = (const float*)d_in[9];

  const int IN = 128, H1 = 4, HID = 64, OUT = 64;
  const int HC1 = H1 * HID;                 // 256
  const int Nn   = in_sizes[0] / IN;        // 50000
  const int E    = in_sizes[1] / 2;         // 800000
  const int Etot = E + Nn;                  // + self loops

  // Workspace layout (floats). Peak ~= N*256*2 + N*16 + Etot*4  (~119 MB).
  float* ws = (float*)d_ws;
  float* h1raw = ws;                               // N*HC1 (reused for all layer-2 scratch)
  float* asrc1 = h1raw + (size_t)Nn * HC1;         // N*H1
  float* adst1 = asrc1 + (size_t)Nn * H1;
  float* m1    = adst1 + (size_t)Nn * H1;
  float* den1  = m1    + (size_t)Nn * H1;
  float* e1    = den1  + (size_t)Nn * H1;          // Etot*H1
  float* h1act = e1    + (size_t)Etot * H1;        // N*HC1
  // Layer-2 scratch aliased into h1raw region (h1raw last read by layer-1 agg):
  float* h2raw = h1raw;                            // N*OUT
  float* asrc2 = h2raw + (size_t)Nn * OUT;         // N
  float* adst2 = asrc2 + (size_t)Nn;
  float* m2    = adst2 + (size_t)Nn;
  float* den2  = m2    + (size_t)Nn;
  float* e2    = den2  + (size_t)Nn;               // Etot

  float* outp = (float*)d_out;                     // N*OUT fp32

  const int BS = 256;                              // 8 waves/block

  // ---- Layer 1 ----
  {
    const int waveTiles = (Nn / 16) * (HC1 / 64);  // 16x64 tile per wave
    const unsigned blocks = (unsigned)((waveTiles + 7) / 8);
    const size_t lds = (size_t)HC1 * (IN + 8) * sizeof(_Float16);   // ~68 KB < 320 KB
    gemm_wmma_f16<<<blocks, BS, lds, stream>>>(x, W1, h1raw, Nn, HC1, IN);
  }
  attn_coef_kernel<<<blocks_for((size_t)Nn * H1, BS), BS, 0, stream>>>(
      h1raw, att_src1, att_dst1, asrc1, adst1, Nn, H1, HID);
  fill_kernel<<<blocks_for((size_t)Nn * H1, BS), BS, 0, stream>>>(m1, -INFINITY, (size_t)Nn * H1);
  fill_kernel<<<blocks_for((size_t)Nn * H1, BS), BS, 0, stream>>>(den1, 0.f, (size_t)Nn * H1);
  fill_kernel<<<blocks_for((size_t)Nn * HC1, BS), BS, 0, stream>>>(h1act, 0.f, (size_t)Nn * HC1);
  edge_logits_max_kernel<<<blocks_for((size_t)Etot * H1, BS), BS, 0, stream>>>(
      ei, E, Etot, asrc1, adst1, e1, m1, H1);
  edge_exp_sum_kernel<<<blocks_for((size_t)Etot * H1, BS), BS, 0, stream>>>(
      ei, E, Etot, m1, e1, den1, H1);
  edge_agg_kernel<<<blocks_for((size_t)Etot * H1 * (HID / 4), BS), BS, 0, stream>>>(
      ei, E, Etot, e1, den1, h1raw, h1act, H1, HID);
  bias_act_kernel<<<blocks_for((size_t)Nn * HC1, BS), BS, 0, stream>>>(
      h1act, b1, HC1, (size_t)Nn * HC1, /*relu=*/1);

  // ---- Layer 2 ----
  {
    const int waveTiles = (Nn / 16) * (OUT / 64);
    const unsigned blocks = (unsigned)((waveTiles + 7) / 8);
    const size_t lds = (size_t)OUT * (HC1 + 8) * sizeof(_Float16);  // ~33 KB
    gemm_wmma_f16<<<blocks, BS, lds, stream>>>(h1act, W2, h2raw, Nn, OUT, HC1);
  }
  attn_coef_kernel<<<blocks_for((size_t)Nn, BS), BS, 0, stream>>>(
      h2raw, att_src2, att_dst2, asrc2, adst2, Nn, 1, OUT);
  fill_kernel<<<blocks_for((size_t)Nn, BS), BS, 0, stream>>>(m2, -INFINITY, (size_t)Nn);
  fill_kernel<<<blocks_for((size_t)Nn, BS), BS, 0, stream>>>(den2, 0.f, (size_t)Nn);
  fill_kernel<<<blocks_for((size_t)Nn * OUT, BS), BS, 0, stream>>>(outp, 0.f, (size_t)Nn * OUT);
  edge_logits_max_kernel<<<blocks_for((size_t)Etot, BS), BS, 0, stream>>>(
      ei, E, Etot, asrc2, adst2, e2, m2, 1);
  edge_exp_sum_kernel<<<blocks_for((size_t)Etot, BS), BS, 0, stream>>>(
      ei, E, Etot, m2, e2, den2, 1);
  edge_agg_kernel<<<blocks_for((size_t)Etot * (OUT / 4), BS), BS, 0, stream>>>(
      ei, E, Etot, e2, den2, h2raw, outp, 1, OUT);
  bias_act_kernel<<<blocks_for((size_t)Nn * OUT, BS), BS, 0, stream>>>(
      outp, b2, OUT, (size_t)Nn * OUT, /*relu=*/0);
}
